// GNNCritic_2826088481167
// MI455X (gfx1250) — matrix operand order repaired and checked by reading, not compiled, not run on hardware
//
#include <hip/hip_runtime.h>
#include <hip/hip_bf16.h>
#include <stdint.h>

// ---------------------------------------------------------------------------
// GNN critic for MI455X (gfx1250, wave32).
// GEMMs run on v_wmma_f32_16x16x32_bf16 (f32 accumulate); edge aggregation is
// an L2-resident f32 atomic scatter (xw/agg are 51MB each, L2 is 192MB).
// ---------------------------------------------------------------------------

typedef __attribute__((ext_vector_type(16))) __bf16 v16bf;
typedef __attribute__((ext_vector_type(8)))  float  v8f;

#define C_IN 128
#define HID  256
#define ACT  20

union ABReg { v16bf v; uint4 u[2]; };

static __device__ __forceinline__ v8f wmma_bf16(const ABReg& a, const ABReg& b, v8f c) {
  // 8 args: (neg_a, A, neg_b, B, c_mod, C, reuse_a, reuse_b)
  return __builtin_amdgcn_wmma_f32_16x16x32_bf16(false, a.v, false, b.v, (short)0, c,
                                                 false, false);
}

// ---------------------------------------------------------------------------
// Init: zero agg, seed deg=1 (self loop), x -> bf16, out[g] = b3.
// ---------------------------------------------------------------------------
__global__ void k_init(const float* __restrict__ x, const float* __restrict__ b3,
                       float* __restrict__ agg, __bf16* __restrict__ xbf,
                       float* __restrict__ deg, float* __restrict__ out,
                       int N, int B) {
  int idx = blockIdx.x * blockDim.x + threadIdx.x;
  int total = N * C_IN;
  if (idx < total) {
    agg[idx] = 0.0f;
    xbf[idx] = (__bf16)x[idx];
  }
  if (idx < N) deg[idx] = 1.0f;     // self-loop contributes 1 to in-degree
  if (idx < B) out[idx] = b3[0];    // fold lin3 bias in up-front
}

// ---------------------------------------------------------------------------
// In-degree accumulation over dst, then dinv = rsqrt(deg).
// ---------------------------------------------------------------------------
__global__ void k_deg(const long long* __restrict__ ei, float* __restrict__ deg, int E) {
  int e = blockIdx.x * blockDim.x + threadIdx.x;
  if (e >= E) return;
  long long d = ei[(size_t)E + e];
  atomicAdd(&deg[d], 1.0f);
}

__global__ void k_rsqrt(float* __restrict__ deg, int N) {
  int i = blockIdx.x * blockDim.x + threadIdx.x;
  if (i >= N) return;
  deg[i] = rsqrtf(deg[i]);          // deg >= 1 always (self loop)
}

// ---------------------------------------------------------------------------
// Pack weights into per-lane WMMA B layout (32x16 bf16 per tile, tile index
// kt*NT+nt; lane l holds column n=l&15, K-half (l>>4)*16..+15, 16 contiguous
// bf16 = 32 bytes per lane).  conv_w is [in,out] (B[k][n]=cw[k*128+n]);
// lin1_w/lin2_w are torch [out,in] (B[k][n]=w[n*K+k]).
// ---------------------------------------------------------------------------
__global__ void k_pack(const float* __restrict__ cw, const float* __restrict__ l1,
                       const float* __restrict__ l2, __bf16* __restrict__ pwc,
                       __bf16* __restrict__ pw1, __bf16* __restrict__ pw2) {
  int gid = blockIdx.x * blockDim.x + threadIdx.x;
  if (gid >= (32 + 64 + 128) * 512) return;
  int t = gid >> 9;          // global tile index
  int e = gid & 511;         // element in tile
  int lane = e >> 4, j = e & 15;
  int kh = lane >> 4, nl = lane & 15;
  if (t < 32) {                       // conv: KT=4, NT=8
    int kt = t >> 3, nt = t & 7;
    int k = kt * 32 + kh * 16 + j, n = nt * 16 + nl;
    pwc[t * 512 + e] = (__bf16)cw[k * C_IN + n];
  } else if (t < 96) {                // lin1: KT=4, NT=16
    int tl = t - 32;
    int kt = tl >> 4, nt = tl & 15;
    int k = kt * 32 + kh * 16 + j, n = nt * 16 + nl;
    pw1[tl * 512 + e] = (__bf16)l1[n * C_IN + k];
  } else {                            // lin2: KT=8, NT=16
    int tl = t - 96;
    int kt = tl >> 4, nt = tl & 15;
    int k = kt * 32 + kh * 16 + j, n = nt * 16 + nl;
    pw2[tl * 512 + e] = (__bf16)l2[n * HID + k];
  }
}

// ---------------------------------------------------------------------------
// xw = x @ conv_w.  One 16-row tile per block, 8 waves = 8 n-tiles, K=128.
// A layout (this code's contiguous k order, as k_pack): lane l -> row m=l&15, K-halves chosen by q=l>>4; row-major
// bf16 makes each half a single 16B load.
// ---------------------------------------------------------------------------
__global__ void k_gemm_conv(const __bf16* __restrict__ xbf, const __bf16* __restrict__ pwc,
                            float* __restrict__ xw) {
  int tile = blockIdx.x;
  int wave = threadIdx.x >> 5;          // n-tile 0..7
  int lane = threadIdx.x & 31;
  int m = lane & 15, q = lane >> 4;
  const __bf16* abase = xbf + ((size_t)tile * 16 + m) * C_IN + q * 16;
  v8f acc = {};
#pragma unroll
  for (int kt = 0; kt < 4; ++kt) {
    ABReg a, b;
    const __bf16* pa = abase + kt * 32;
    a.u[0] = *(const uint4*)pa;
    a.u[1] = *(const uint4*)(pa + 8);
    const __bf16* pb = pwc + ((kt * 8 + wave) * 512 + lane * 16);
    b.u[0] = *(const uint4*)pb;
    b.u[1] = *(const uint4*)(pb + 8);
    acc = wmma_bf16(a, b, acc);
  }
#pragma unroll
  for (int r = 0; r < 8; ++r)
    xw[((size_t)tile * 16 + r + 8 * q) * C_IN + wave * 16 + m] = acc[r];
}

// ---------------------------------------------------------------------------
// Edge scatter: one wave per edge, 4 channels per lane (float4 gather,
// 4x global_atomic_add_f32 scatter).  L2-resident by construction.
// ---------------------------------------------------------------------------
__global__ void k_scatter(const long long* __restrict__ ei, const float* __restrict__ xw,
                          const float* __restrict__ dinv, float* __restrict__ agg, int E) {
  int widx = (blockIdx.x * blockDim.x + threadIdx.x) >> 5;
  if (widx >= E) return;
  int lane = threadIdx.x & 31;
  long long s = ei[widx];
  long long d = ei[(size_t)E + widx];
  if (widx + 2048 < E) __builtin_prefetch(ei + widx + 2048, 0, 0);
  float w = dinv[s] * dinv[d];
  float4 xv = *(const float4*)(xw + (size_t)s * C_IN + lane * 4);
  float* ap = agg + (size_t)d * C_IN + lane * 4;
  atomicAdd(ap + 0, xv.x * w);
  atomicAdd(ap + 1, xv.y * w);
  atomicAdd(ap + 2, xv.z * w);
  atomicAdd(ap + 3, xv.w * w);
}

// ---------------------------------------------------------------------------
// Self-loop + bias + relu + residual -> h (bf16 for the MLP GEMMs).
// ---------------------------------------------------------------------------
__global__ void k_post(const float* __restrict__ x, const float* __restrict__ xw,
                       const float* __restrict__ agg, const float* __restrict__ dinv,
                       const float* __restrict__ cb, __bf16* __restrict__ hbf, int N) {
  int idx = blockIdx.x * blockDim.x + threadIdx.x;
  if (idx >= N * C_IN) return;
  int n = idx >> 7, c = idx & 127;
  float di = dinv[n];
  float v = agg[idx] + xw[idx] * di * di + cb[c];
  hbf[idx] = (__bf16)(fmaxf(v, 0.0f) + x[idx]);
}

// ---------------------------------------------------------------------------
// Fused MLP: 16 rows per block, 4 waves (4 n-tiles each).
//  h1 = relu(h @ W1^T + b1)  (K=128, N=256)  -> LDS (stride 264 bf16)
//  h2 = relu(h1 @ W2^T + b2) (K=256, N=256)
//  s[m] = dot(h2[m], w3); out[row/20] += s[m]   (lin3 folded per-row)
// ---------------------------------------------------------------------------
__global__ void k_mlp(const __bf16* __restrict__ hbf, const __bf16* __restrict__ pw1,
                      const float* __restrict__ b1, const __bf16* __restrict__ pw2,
                      const float* __restrict__ b2, const float* __restrict__ w3,
                      float* __restrict__ out) {
  __shared__ __bf16 h1s[16 * 264];   // 528B row stride: 16B aligned, conflict-free
  int tile = blockIdx.x;
  int wave = threadIdx.x >> 5;
  int lane = threadIdx.x & 31;
  int m = lane & 15, q = lane >> 4;

  // ---- lin1: A from global h, 4 n-tiles per wave, K=128 ----
  v8f acc[4] = {};
  const __bf16* abase = hbf + ((size_t)tile * 16 + m) * C_IN + q * 16;
#pragma unroll
  for (int kt = 0; kt < 4; ++kt) {
    ABReg a;
    const __bf16* pa = abase + kt * 32;
    a.u[0] = *(const uint4*)pa;
    a.u[1] = *(const uint4*)(pa + 8);
#pragma unroll
    for (int t = 0; t < 4; ++t) {
      int nt = wave * 4 + t;
      ABReg b;
      const __bf16* pb = pw1 + ((kt * 16 + nt) * 512 + lane * 16);
      b.u[0] = *(const uint4*)pb;
      b.u[1] = *(const uint4*)(pb + 8);
      acc[t] = wmma_bf16(a, b, acc[t]);
    }
  }
#pragma unroll
  for (int t = 0; t < 4; ++t) {
    int nt = wave * 4 + t;
    float bias = b1[nt * 16 + m];
#pragma unroll
    for (int r = 0; r < 8; ++r) {
      float v = fmaxf(acc[t][r] + bias, 0.0f);
      h1s[(r + 8 * q) * 264 + nt * 16 + m] = (__bf16)v;
    }
  }
  __syncthreads();

  // ---- lin2: A from LDS (b128 ds loads), K=256 ----
  v8f acc2[4] = {};
#pragma unroll
  for (int kt = 0; kt < 8; ++kt) {
    ABReg a;
    const __bf16* pa = h1s + m * 264 + kt * 32 + q * 16;
    a.u[0] = *(const uint4*)pa;
    a.u[1] = *(const uint4*)(pa + 8);
#pragma unroll
    for (int t = 0; t < 4; ++t) {
      int nt = wave * 4 + t;
      ABReg b;
      const __bf16* pb = pw2 + ((kt * 16 + nt) * 512 + lane * 16);
      b.u[0] = *(const uint4*)pb;
      b.u[1] = *(const uint4*)(pb + 8);
      acc2[t] = wmma_bf16(a, b, acc2[t]);
    }
  }

  // ---- epilogue: relu, fold lin3 dot, half-wave reduce, segment atomicAdd ----
  float part[8] = {0, 0, 0, 0, 0, 0, 0, 0};
#pragma unroll
  for (int t = 0; t < 4; ++t) {
    int nt = wave * 4 + t;
    float bias = b2[nt * 16 + m];
    float wv = w3[nt * 16 + m];
#pragma unroll
    for (int r = 0; r < 8; ++r)
      part[r] += fmaxf(acc2[t][r] + bias, 0.0f) * wv;
  }
#pragma unroll
  for (int r = 0; r < 8; ++r) {
    part[r] += __shfl_xor(part[r], 1);
    part[r] += __shfl_xor(part[r], 2);
    part[r] += __shfl_xor(part[r], 4);
    part[r] += __shfl_xor(part[r], 8);
  }
  if (m == 0) {
    int rb = tile * 16 + 8 * q;
#pragma unroll
    for (int r = 0; r < 8; ++r)
      atomicAdd(&out[(rb + r) / ACT], part[r]);
  }
}

// ---------------------------------------------------------------------------
extern "C" void kernel_launch(void* const* d_in, const int* in_sizes, int n_in,
                              void* d_out, int out_size, void* d_ws, size_t ws_size,
                              hipStream_t stream) {
  (void)n_in; (void)out_size; (void)ws_size;
  const float*     x   = (const float*)d_in[0];
  const long long* ei  = (const long long*)d_in[1];   // int64 [2,E]
  const float*     cw  = (const float*)d_in[2];
  const float*     cb  = (const float*)d_in[3];
  const float*     l1w = (const float*)d_in[4];
  const float*     l1b = (const float*)d_in[5];
  const float*     l2w = (const float*)d_in[6];
  const float*     l2b = (const float*)d_in[7];
  const float*     l3w = (const float*)d_in[8];
  const float*     l3b = (const float*)d_in[9];
  float* out = (float*)d_out;

  int N = in_sizes[0] / C_IN;
  int E = in_sizes[1] / 2;
  int B = N / ACT;

  char* ws = (char*)d_ws;
  size_t sz_xbf = (size_t)N * C_IN * 2;   // also reused as hbf after GEMM1
  size_t sz_f32 = (size_t)N * C_IN * 4;
  __bf16* xbf  = (__bf16*)ws;
  float*  xw   = (float*)(ws + sz_xbf);
  float*  agg  = (float*)(ws + sz_xbf + sz_f32);
  float*  dinv = (float*)(ws + sz_xbf + 2 * sz_f32);
  char*   pwb  = ws + sz_xbf + 2 * sz_f32 + (size_t)N * 4;
  __bf16* pwc = (__bf16*)pwb;                    // 32 tiles * 1KB
  __bf16* pw1 = (__bf16*)(pwb + 32 * 1024);      // 64 tiles
  __bf16* pw2 = (__bf16*)(pwb + 96 * 1024);      // 128 tiles

  int totNC = N * C_IN;
  k_init<<<(totNC + 255) / 256, 256, 0, stream>>>(x, l3b, agg, xbf, dinv, out, N, B);
  k_deg<<<(E + 255) / 256, 256, 0, stream>>>(ei, dinv, E);
  k_rsqrt<<<(N + 255) / 256, 256, 0, stream>>>(dinv, N);
  k_pack<<<(224 * 512) / 256, 256, 0, stream>>>(cw, l1w, l2w, pwc, pw1, pw2);
  k_gemm_conv<<<N / 16, 256, 0, stream>>>(xbf, pwc, xw);
  k_scatter<<<(E + 7) / 8, 256, 0, stream>>>(ei, xw, dinv, agg, E);
  k_post<<<(totNC + 255) / 256, 256, 0, stream>>>(x, xw, agg, dinv, cb, xbf /*hbf*/, N);
  k_mlp<<<N / 16, 128, 0, stream>>>(xbf /*hbf*/, pw1, l1b, pw2, l2b, l3w, out);
}
